// BiLSTM_CRF_34918084117004
// MI455X (gfx1250) — matrix operand order repaired and checked by reading, not compiled, not run on hardware
//
#include <hip/hip_runtime.h>
#include <hip/hip_bf16.h>
#include <cstdint>

// BiLSTM-CRF for MI455X (gfx1250, wave32).
//
//  1) f32->bf16 conversions (bf16 WMMA inputs, f32 accumulate).
//  2) xg = X @ W_ih^T + (b_ih+b_hh): 128x128 block-tiled WMMA GEMM with
//     double-buffered LDS staging (v_wmma_f32_16x16x32_bf16, ds_load frags).
//  3) Recurrence: persistent kernel, 32 blocks x 2 directions; each block pins
//     its 128 gate rows of W_hh (256 KB) in LDS via the Tensor Data Mover
//     (tensor_load_to_lds + s_wait_tensorcnt), cell state in registers,
//     device-scope barrier per step, double-buffered h.
//  4) feats projection + single-wave Viterbi.

#define T_LEN 8192
#define E_IN  512
#define HN    1024
#define G4H   4096
#define LBL   5
#define NB    32      // blocks per direction in the recurrent kernel
#define CHUNK 32      // hidden units per block
#define RPB   128     // gate rows per block = 4*CHUNK
#define NEGV  -10000.0f

// GEMM tiling
#define MT   128
#define NT   128
#define KT   32
#define LDST 40       // padded LDS row stride in bf16 (80B -> conflict-free frags)

typedef __attribute__((ext_vector_type(16))) __bf16 v16bf;
typedef __attribute__((ext_vector_type(8)))  __bf16 v8bf;
typedef __attribute__((ext_vector_type(8)))  float  v8f;
typedef __attribute__((ext_vector_type(4)))  unsigned int v4u;
typedef __attribute__((ext_vector_type(8)))  int v8i;
typedef __attribute__((ext_vector_type(4)))  int v4i;

__device__ __forceinline__ float sigmoidf_(float x) { return 1.0f / (1.0f + __expf(-x)); }

// ---------------------------------------------------------------- utilities
__global__ void f32_to_bf16_kernel(const float* __restrict__ in,
                                   __hip_bfloat16* __restrict__ out, unsigned n) {
  unsigned i = blockIdx.x * blockDim.x + threadIdx.x;
  if (i < n) out[i] = __float2bfloat16(in[i]);
}

__global__ void zero_u32_kernel(unsigned* __restrict__ p, unsigned n) {
  unsigned i = blockIdx.x * blockDim.x + threadIdx.x;
  if (i < n) p[i] = 0u;
}

// ------------------------------------------------ WMMA GEMM: C = A @ W^T + b
// A: [M,K] bf16, W: [N,K] bf16, C: [M,N] bf16. Block = 256 thr = 8 waves,
// computes a 128x128 tile; waves in a 2x4 grid, each wave 64x32 (4x2 subtiles).
// K-steps of 32 double-buffered through LDS (16 KB global traffic / 64 WMMAs).
// Fragment layouts per CDNA5 ISA 7.12.2:
//   A (16x32): lanes0-15 row=lane, K={0..7,16..23}; lanes16-31 K={8..15,24..31}
//   B (32x16): lanes0-15 col=lane, K={0..7,16..23}; lanes16-31 K={8..15,24..31}
//   C/D f32:   vgpr r, lanes0-15 M=r, lanes16-31 M=8+r; N=lane&15
__global__ __launch_bounds__(256)
void wmma_gemm_xg(const __hip_bfloat16* __restrict__ A,
                  const __hip_bfloat16* __restrict__ W,
                  const float* __restrict__ b0,
                  const float* __restrict__ b1,
                  __hip_bfloat16* __restrict__ C,
                  int M, int N, int K) {
  __shared__ __align__(16) __hip_bfloat16 As[2][MT * LDST];   // 2 x 10 KB
  __shared__ __align__(16) __hip_bfloat16 Bs[2][NT * LDST];   // 2 x 10 KB

  const int tid  = threadIdx.x;
  const int lane = tid & 31;
  const int wave = tid >> 5;
  const int m0   = blockIdx.x * MT;
  const int n0   = blockIdx.y * NT;
  const int wrow = wave >> 2;          // 0..1 : M offset wrow*64
  const int wcol = wave & 3;           // 0..3 : N offset wcol*32
  const int hi   = lane >> 4;
  const int l15  = lane & 15;

  v8f acc[4][2] = {};

  // Stage one 128x32 A tile and 128x32 W tile into LDS buffer p.
  // 512 16B chunks per tile, 2 per thread; chunk ci -> row=ci>>2, col8=ci&3.
  auto stage = [&](int p, int kb) {
#pragma unroll
    for (int t = 0; t < 2; ++t) {
      int ci  = tid + t * 256;
      int row = ci >> 2;
      int c   = ci & 3;
      *(uint4*)&As[p][row * LDST + c * 8] =
          *(const uint4*)(A + (size_t)(m0 + row) * K + kb + c * 8);
      *(uint4*)&Bs[p][row * LDST + c * 8] =
          *(const uint4*)(W + (size_t)(n0 + row) * K + kb + c * 8);
      if (kb + KT < K) {               // GL2 prefetch of the tile after next
        __builtin_prefetch(A + (size_t)(m0 + row) * K + kb + KT + c * 8, 0, 1);
        __builtin_prefetch(W + (size_t)(n0 + row) * K + kb + KT + c * 8, 0, 1);
      }
    }
  };

  stage(0, 0);
  int p = 0;
  for (int kb = 0; kb < K; kb += KT) {
    __syncthreads();
    if (kb + KT < K) stage(p ^ 1, kb + KT);

    v16bf af[4], bfr[2];
#pragma unroll
    for (int i = 0; i < 4; ++i) {
      const __hip_bfloat16* src = &As[p][(wrow * 64 + i * 16 + l15) * LDST];
      v8bf lo = *(const v8bf*)(src + hi * 8);
      v8bf hi8 = *(const v8bf*)(src + 16 + hi * 8);
      af[i] = __builtin_shufflevector(lo, hi8,
          0, 1, 2, 3, 4, 5, 6, 7, 8, 9, 10, 11, 12, 13, 14, 15);
    }
#pragma unroll
    for (int j = 0; j < 2; ++j) {
      const __hip_bfloat16* src = &Bs[p][(wcol * 32 + j * 16 + l15) * LDST];
      v8bf lo = *(const v8bf*)(src + hi * 8);
      v8bf hi8 = *(const v8bf*)(src + 16 + hi * 8);
      bfr[j] = __builtin_shufflevector(lo, hi8,
          0, 1, 2, 3, 4, 5, 6, 7, 8, 9, 10, 11, 12, 13, 14, 15);
    }
#pragma unroll
    for (int i = 0; i < 4; ++i)
#pragma unroll
      for (int j = 0; j < 2; ++j)
        acc[i][j] = __builtin_amdgcn_wmma_f32_16x16x32_bf16(
            false, af[i], false, bfr[j], (short)0, acc[i][j], false, false);
    p ^= 1;
  }

#pragma unroll
  for (int j = 0; j < 2; ++j) {
    const int n = n0 + wcol * 32 + j * 16 + l15;
    const float bias = b0[n] + b1[n];
#pragma unroll
    for (int i = 0; i < 4; ++i) {
      const int mb = m0 + wrow * 64 + i * 16 + hi * 8;
#pragma unroll
      for (int r = 0; r < 8; ++r)
        C[(size_t)(mb + r) * N + n] = __float2bfloat16(acc[i][j][r] + bias);
    }
  }
}

// ---------------------------------------------------- persistent recurrence
__device__ __forceinline__ void grid_barrier(unsigned* cnt, unsigned* gen, unsigned nb) {
  __syncthreads();
  if (threadIdx.x == 0) {
    volatile unsigned* vgen = (volatile unsigned*)gen;
    unsigned g = *vgen;
    __threadfence();
    unsigned a = atomicAdd(cnt, 1u);
    if (a == nb - 1u) {
      *(volatile unsigned*)cnt = 0u;
      __threadfence();
      atomicAdd(gen, 1u);
    } else {
      while (*vgen == g) { __builtin_amdgcn_s_sleep(2); }
    }
  }
  __syncthreads();
}

// TDM preload: one 2D-tile descriptor per gate strip (32 rows x 1024 bf16,
// row stride H), per CDNA5 ISA. Issued by wave 0,
// completed with s_wait_tensorcnt 0.
#if defined(__has_builtin)
#if __has_builtin(__builtin_amdgcn_tensor_load_to_lds)
#define HAVE_TDM 1
#endif
#endif

__device__ __forceinline__ void tdm_load_gate_strip(
    const __hip_bfloat16* gsrc, unsigned lds_byte_off) {
#ifdef HAVE_TDM
  const unsigned long long ga = (unsigned long long)(uintptr_t)(const void*)gsrc;
  // D# group 0: count=1 | lds_addr | global_addr | type=2
  v4u g0 = { 1u,                                  // count=1 (valid descriptor)
             lds_byte_off,                        // lds_addr (bytes)
             (unsigned)(ga & 0xffffffffu),        // global_addr[31:0]
             (unsigned)(ga >> 32) | (2u << 30) }; // global_addr[56:32] | type=2
  // D# group 1: data_size=2B; tensor_dim0=H, tensor_dim1=4H;
  //             tile_dim0=H, tile_dim1=32; tensor_dim0_stride=H
  v8i g1 = { (int)(1u << 16),                 // [17:16] data_size = 1 (2 bytes)
             0,                               // atomic barrier addr = 0
             (int)((unsigned)HN << 16),       // tensor_dim0[15:0] << 16
             (int)(((unsigned)G4H & 0xffffu) << 16), // dim0 hi | tensor_dim1 lo
             (int)((unsigned)HN << 16),       // tensor_dim1 hi | tile_dim0 = H
             (int)32,                         // tile_dim1 = 32 | tile_dim2 = 0
             (int)HN,                         // tensor_dim0_stride[31:0] = H
             0 };                             // stride hi | tensor_dim1_stride
  v4i gz = { 0, 0, 0, 0 };                    // groups 2/3 unused (2D tensor)
#if __clang_major__ >= 23
  v8i gz8 = { 0, 0, 0, 0, 0, 0, 0, 0 };
  __builtin_amdgcn_tensor_load_to_lds(g0, g1, gz, gz, gz8, 0);
#else
  __builtin_amdgcn_tensor_load_to_lds(g0, g1, gz, gz, 0);
#endif
#else
  (void)gsrc; (void)lds_byte_off;
#endif
}

// grid = (NB, 2): blockIdx.y = direction (0=fwd, 1=bwd). 256 threads.
// Dynamic LDS: [RPB][HN] bf16 weights (256KB) + h stage (4KB) + gates (512B).
__global__ __launch_bounds__(256)
void lstm_recurrent(const __hip_bfloat16* __restrict__ xg_f,   // [T,4H]
                    const __hip_bfloat16* __restrict__ xg_r,
                    const __hip_bfloat16* __restrict__ whh_f,  // [4H,H]
                    const __hip_bfloat16* __restrict__ whh_r,
                    float* __restrict__ hbuf,                  // [2 dir][2 phase][H]
                    unsigned* __restrict__ bars,               // [2 dir][cnt,gen]
                    float* __restrict__ hs_out)                // [T, 2H] time-aligned
{
  extern __shared__ char smem[];
  __hip_bfloat16* wl  = (__hip_bfloat16*)smem;                 // [RPB][HN]
  float*          hsh = (float*)(smem + (size_t)RPB * HN * 2); // [HN]
  float*          gsh = hsh + HN;                              // [RPB]

  const int dir = blockIdx.y;
  const int cb  = blockIdx.x * CHUNK;
  const __hip_bfloat16* xg  = dir ? xg_r  : xg_f;
  const __hip_bfloat16* whh = dir ? whh_r : whh_f;

  // Pin this block's 128 gate rows of W_hh in LDS for the whole sequence.
#ifdef HAVE_TDM
  if (threadIdx.x < 32) {              // wave 0 drives the Tensor Data Mover
    const unsigned wlOff = (unsigned)(uintptr_t)(void*)wl;
#pragma unroll
    for (int g = 0; g < 4; ++g)
      tdm_load_gate_strip(whh + ((size_t)g * HN + cb) * HN,
                          wlOff + (unsigned)g * (CHUNK * HN * 2));
    __builtin_amdgcn_s_wait_tensorcnt(0);
  }
#else
  for (int i = threadIdx.x; i < RPB * (HN / 8); i += blockDim.x) {
    int rr = i >> 7;
    int j  = i & 127;
    size_t grow = (size_t)(rr >> 5) * HN + cb + (rr & 31);
    ((uint4*)wl)[(size_t)rr * (HN / 8) + j] = ((const uint4*)(whh + grow * HN))[j];
  }
#endif
  __syncthreads();

  const int r    = threadIdx.x >> 1;   // gate row 0..127 (2 threads per row)
  const int half = threadIdx.x & 1;
  const __hip_bfloat162* wrow =
      (const __hip_bfloat162*)(wl + (size_t)r * HN + half * (HN / 2));
  const size_t gidx = (size_t)(r >> 5) * HN + cb + (r & 31);
  float creg = 0.0f;                   // cell state (threads 0..31 only)

  for (int s = 0; s < T_LEN; ++s) {
    const int tt = dir ? (T_LEN - 1 - s) : s;
    const int rp = s & 1, wp = (s + 1) & 1;

    ((float4*)hsh)[threadIdx.x] =
        ((const float4*)(hbuf + (size_t)(dir * 2 + rp) * HN))[threadIdx.x];
    __syncthreads();

    const float2* hp = (const float2*)(hsh + half * (HN / 2));
    float sum = 0.0f;
#pragma unroll 8
    for (int k = 0; k < HN / 4; ++k) {
      __hip_bfloat162 wv = wrow[k];
      float2 hv = hp[k];
      sum = fmaf(__low2float(wv),  hv.x, sum);
      sum = fmaf(__high2float(wv), hv.y, sum);
    }
    sum += __shfl_xor(sum, 1, 32);
    if (half == 0)
      gsh[r] = sum + __bfloat162float(xg[(size_t)tt * G4H + gidx]);
    __syncthreads();

    if (threadIdx.x < CHUNK) {         // LSTM cell, torch gate order i,f,g,o
      int u = threadIdx.x;
      float gi = gsh[u], gf = gsh[CHUNK + u], gg = gsh[2 * CHUNK + u],
            go = gsh[3 * CHUNK + u];
      float c = sigmoidf_(gf) * creg + sigmoidf_(gi) * tanhf(gg);
      creg = c;
      float h = sigmoidf_(go) * tanhf(c);
      hbuf[(size_t)(dir * 2 + wp) * HN + cb + u] = h;
      hs_out[(size_t)tt * (2 * HN) + (size_t)dir * HN + cb + u] = h;
    }
    __threadfence();
    grid_barrier(bars + dir * 2, bars + dir * 2 + 1, NB);
  }
}

// -------------------------------------------- feats = H1 @ W_out^T + b_out
__global__ __launch_bounds__(256)
void proj_kernel(const float* __restrict__ h1, const float* __restrict__ wout,
                 const float* __restrict__ bout, float* __restrict__ feats) {
  __shared__ float red[256];
  const int t = blockIdx.x;
  const float* hrow = h1 + (size_t)t * (2 * HN);
  for (int l = 0; l < LBL; ++l) {
    float s = 0.0f;
    for (int j = threadIdx.x; j < 2 * HN; j += 256)
      s = fmaf(hrow[j], wout[(size_t)l * (2 * HN) + j], s);
    red[threadIdx.x] = s;
    __syncthreads();
    for (int st = 128; st > 0; st >>= 1) {
      if (threadIdx.x < st) red[threadIdx.x] += red[threadIdx.x + st];
      __syncthreads();
    }
    if (threadIdx.x == 0) feats[(size_t)t * LBL + l] = red[0] + bout[l];
    __syncthreads();
  }
}

// ------------------------------------------------------- single-wave Viterbi
__global__ void viterbi_kernel(const float* __restrict__ feats,
                               const float* __restrict__ trans,
                               int* __restrict__ bps, float* __restrict__ out) {
  const int lane = threadIdx.x;
  float fv = (lane == 3 /*START*/) ? 0.0f : NEGV;
  for (int t = 0; t < T_LEN; ++t) {
    float best = -3.4e38f; int arg = 0;
    for (int p = 0; p < LBL; ++p) {
      float prev = __shfl(fv, p, 32);
      float s = prev + ((lane < LBL) ? trans[lane * LBL + p] : 0.0f);
      if (s > best) { best = s; arg = p; }
    }
    fv = best + ((lane < LBL) ? feats[(size_t)t * LBL + lane] : 0.0f);
    if (lane < LBL) bps[t * LBL + lane] = arg;
  }
  float term = fv + ((lane < LBL) ? trans[4 /*STOP*/ * LBL + lane] : -3.4e38f);
  float bestS = -3.4e38f; int bestL = 0;
  for (int p = 0; p < LBL; ++p) {
    float s = __shfl(term, p, 32);
    if (s > bestS) { bestS = s; bestL = p; }
  }
  if (lane == 0) {
    out[0] = bestS;
    int lbl = bestL;
    for (int t = T_LEN - 1; t >= 0; --t) {
      out[1 + t] = (float)lbl;
      lbl = bps[t * LBL + lbl];
    }
  }
}

// ------------------------------------------------------------------- driver
extern "C" void kernel_launch(void* const* d_in, const int* in_sizes, int n_in,
                              void* d_out, int out_size, void* d_ws, size_t ws_size,
                              hipStream_t stream) {
  (void)in_sizes; (void)n_in; (void)out_size; (void)ws_size;
  const float* hours    = (const float*)d_in[0];
  const float* w_ih_l0  = (const float*)d_in[1];
  const float* w_hh_l0  = (const float*)d_in[2];
  const float* b_ih_l0  = (const float*)d_in[3];
  const float* b_hh_l0  = (const float*)d_in[4];
  const float* w_ih_l0r = (const float*)d_in[5];
  const float* w_hh_l0r = (const float*)d_in[6];
  const float* b_ih_l0r = (const float*)d_in[7];
  const float* b_hh_l0r = (const float*)d_in[8];
  const float* w_ih_l1  = (const float*)d_in[9];
  const float* w_hh_l1  = (const float*)d_in[10];
  const float* b_ih_l1  = (const float*)d_in[11];
  const float* b_hh_l1  = (const float*)d_in[12];
  const float* w_ih_l1r = (const float*)d_in[13];
  const float* w_hh_l1r = (const float*)d_in[14];
  const float* b_ih_l1r = (const float*)d_in[15];
  const float* b_hh_l1r = (const float*)d_in[16];
  const float* w_out    = (const float*)d_in[17];
  const float* b_out    = (const float*)d_in[18];
  const float* trans    = (const float*)d_in[19];
  float* out = (float*)d_out;

  char* ws = (char*)d_ws;
  size_t off = 0;
  auto take = [&](size_t bytes) -> char* {
    char* p = ws + off;
    off += (bytes + 255) & ~(size_t)255;
    return p;
  };
  __hip_bfloat16* hoursBf = (__hip_bfloat16*)take((size_t)T_LEN * E_IN * 2);
  __hip_bfloat16* wih0f   = (__hip_bfloat16*)take((size_t)G4H * E_IN * 2);
  __hip_bfloat16* wih0r   = (__hip_bfloat16*)take((size_t)G4H * E_IN * 2);
  __hip_bfloat16* wih1f   = (__hip_bfloat16*)take((size_t)G4H * 2 * HN * 2);
  __hip_bfloat16* wih1r   = (__hip_bfloat16*)take((size_t)G4H * 2 * HN * 2);
  __hip_bfloat16* whh0f   = (__hip_bfloat16*)take((size_t)G4H * HN * 2);
  __hip_bfloat16* whh0r   = (__hip_bfloat16*)take((size_t)G4H * HN * 2);
  __hip_bfloat16* whh1f   = (__hip_bfloat16*)take((size_t)G4H * HN * 2);
  __hip_bfloat16* whh1r   = (__hip_bfloat16*)take((size_t)G4H * HN * 2);
  __hip_bfloat16* xgA     = (__hip_bfloat16*)take((size_t)T_LEN * G4H * 2);
  __hip_bfloat16* xgB     = (__hip_bfloat16*)take((size_t)T_LEN * G4H * 2);
  float*          h0cat   = (float*)take((size_t)T_LEN * 2 * HN * 4);
  float*          h1cat   = (float*)take((size_t)T_LEN * 2 * HN * 4);
  __hip_bfloat16* h0catBf = (__hip_bfloat16*)take((size_t)T_LEN * 2 * HN * 2);
  float*          hbuf    = (float*)take((size_t)2 * 2 * HN * 4);   // + bars contiguous
  unsigned*       bars    = (unsigned*)take(4 * sizeof(unsigned));
  float*          feats   = (float*)take((size_t)T_LEN * LBL * 4);
  int*            bps     = (int*)take((size_t)T_LEN * LBL * 4);

  auto cvt = [&](const float* src, __hip_bfloat16* dst, size_t n) {
    f32_to_bf16_kernel<<<dim3((unsigned)((n + 255) / 256)), 256, 0, stream>>>(
        src, dst, (unsigned)n);
  };
  auto zero_state = [&]() {  // zero hbuf (4096 u32) + bars (4 u32), contiguous
    zero_u32_kernel<<<dim3((4 * HN + 4 + 255) / 256), 256, 0, stream>>>(
        (unsigned*)hbuf, 4 * HN + 4);
  };

  // 1) bf16 conversions
  cvt(hours,    hoursBf, (size_t)T_LEN * E_IN);
  cvt(w_ih_l0,  wih0f,   (size_t)G4H * E_IN);
  cvt(w_ih_l0r, wih0r,   (size_t)G4H * E_IN);
  cvt(w_hh_l0,  whh0f,   (size_t)G4H * HN);
  cvt(w_hh_l0r, whh0r,   (size_t)G4H * HN);
  cvt(w_ih_l1,  wih1f,   (size_t)G4H * 2 * HN);
  cvt(w_ih_l1r, wih1r,   (size_t)G4H * 2 * HN);
  cvt(w_hh_l1,  whh1f,   (size_t)G4H * HN);
  cvt(w_hh_l1r, whh1r,   (size_t)G4H * HN);

  const dim3 gemmGrid(T_LEN / MT, G4H / NT);
  const size_t smem = (size_t)RPB * HN * 2 + (size_t)(HN + RPB) * 4;
  (void)hipFuncSetAttribute((const void*)&lstm_recurrent,
                            hipFuncAttributeMaxDynamicSharedMemorySize, (int)smem);

  // 2) layer 0: xg GEMMs (K=512) then persistent bidirectional recurrence
  wmma_gemm_xg<<<gemmGrid, 256, 0, stream>>>(hoursBf, wih0f, b_ih_l0, b_hh_l0,
                                             xgA, T_LEN, G4H, E_IN);
  wmma_gemm_xg<<<gemmGrid, 256, 0, stream>>>(hoursBf, wih0r, b_ih_l0r, b_hh_l0r,
                                             xgB, T_LEN, G4H, E_IN);
  zero_state();
  lstm_recurrent<<<dim3(NB, 2), 256, smem, stream>>>(xgA, xgB, whh0f, whh0r,
                                                     hbuf, bars, h0cat);

  // 3) layer 1: concat output -> bf16, xg GEMMs (K=2048), recurrence
  cvt(h0cat, h0catBf, (size_t)T_LEN * 2 * HN);
  wmma_gemm_xg<<<gemmGrid, 256, 0, stream>>>(h0catBf, wih1f, b_ih_l1, b_hh_l1,
                                             xgA, T_LEN, G4H, 2 * HN);
  wmma_gemm_xg<<<gemmGrid, 256, 0, stream>>>(h0catBf, wih1r, b_ih_l1r, b_hh_l1r,
                                             xgB, T_LEN, G4H, 2 * HN);
  zero_state();
  lstm_recurrent<<<dim3(NB, 2), 256, smem, stream>>>(xgA, xgB, whh1f, whh1r,
                                                     hbuf, bars, h1cat);

  // 4) projection + Viterbi
  proj_kernel<<<dim3(T_LEN), 256, 0, stream>>>(h1cat, w_out, b_out, feats);
  viterbi_kernel<<<1, 32, 0, stream>>>(feats, trans, bps, out);
}
